// SageConvLayer_19885698580662
// MI455X (gfx1250) — matrix operand order repaired and checked by reading, not compiled, not run on hardware
//
#include <hip/hip_runtime.h>
#include <stdint.h>

// Problem constants (from reference)
#define NN 16384
#define DD 128

typedef __attribute__((ext_vector_type(16))) __bf16 v16bf;
typedef __attribute__((ext_vector_type(8)))  float  v8f;

// ---------------------------------------------------------------------------
// Kernel 1: neigh_feats [N][128] fp32 -> neighT [128][N] bf16 (tiled transpose)
// ---------------------------------------------------------------------------
__global__ __launch_bounds__(256) void k_transpose_bf16(
    const float* __restrict__ src, __bf16* __restrict__ dst) {
  __shared__ float tile[32][33];
  const int n0 = blockIdx.x * 32;
  const int d0 = blockIdx.y * 32;
  const int tx = threadIdx.x;  // 0..31
  const int ty = threadIdx.y;  // 0..7
#pragma unroll
  for (int k = 0; k < 4; ++k) {
    const int n = n0 + ty + k * 8;
    tile[ty + k * 8][tx] = src[(size_t)n * DD + d0 + tx];
  }
  __syncthreads();
#pragma unroll
  for (int k = 0; k < 4; ++k) {
    const int d = d0 + ty + k * 8;
    dst[(size_t)d * NN + n0 + tx] = (__bf16)tile[tx][ty + k * 8];
  }
}

// ---------------------------------------------------------------------------
// Kernel 2: features -> bf16 (row-major), W [128][256] -> bf16 (row-major)
// ---------------------------------------------------------------------------
__global__ __launch_bounds__(256) void k_cvt_bf16(
    const float* __restrict__ feat, const float* __restrict__ W,
    __bf16* __restrict__ featbf, __bf16* __restrict__ wbf) {
  const int t = blockIdx.x * 256 + threadIdx.x;
  const int NF = NN * DD;
  if (t < NF) {
    featbf[t] = (__bf16)feat[t];
  } else {
    const int t2 = t - NF;
    if (t2 < DD * 2 * DD) wbf[t2] = (__bf16)W[t2];
  }
}

// ---------------------------------------------------------------------------
// Kernel 3: agg = (adj @ neigh) / (rowsum(adj)+1), fused rowsum.
// 64 rows per WG (4 waves x 16 rows x 128 cols) -> 256 WGs for occupancy.
// A: per-lane global fp32 loads -> (__bf16) packs, exactly in the WMMA
//    16-bit A layout (lanes 0-15: K {0..7,16..23}, lanes 16-31: K {8..15,24..31}).
// B: 64x128 bf16 panel staged in LDS (double buffered, padded stride).
// ---------------------------------------------------------------------------
#define KB   64
#define BSTR 72  // padded LDS row stride (ushorts): 144B, conflict-free
#define NWAVE 4  // waves per workgroup

__global__ __launch_bounds__(32 * NWAVE) void k_agg_gemm(
    const float* __restrict__ adj, const __bf16* __restrict__ neighT,
    __bf16* __restrict__ aggbf) {
  __shared__ __align__(16) uint16_t Bs[2][DD * BSTR];  // 2 x 18 KB

  const int tid  = threadIdx.x;
  const int wave = tid >> 5;
  const int lane = tid & 31;
  const int ll   = lane & 15;
  const int g    = lane >> 4;
  const int rowbase = blockIdx.x * (16 * NWAVE) + wave * 16;
  const int row     = rowbase + ll;
  const float* arow = adj + (size_t)row * NN;

  v8f acc[8];
#pragma unroll
  for (int i = 0; i < 8; ++i)
#pragma unroll
    for (int j = 0; j < 8; ++j) acc[i][j] = 0.0f;

  auto stage = [&](int buf, int kb) {
#pragma unroll
    for (int j = 0; j < 1024 / (32 * NWAVE); ++j) {
      const int c    = tid + (32 * NWAVE) * j;  // 0..1023 16-byte chunks
      const int d    = c >> 3;
      const int part = c & 7;
      const uint4 v =
          *(const uint4*)((const uint16_t*)neighT + (size_t)d * NN + kb + part * 8);
      *(uint4*)&Bs[buf][d * BSTR + part * 8] = v;
    }
  };

  float rowsum = 0.0f;
  stage(0, 0);
  __syncthreads();

  const int NIT = NN / KB;  // 256
  for (int it = 0; it < NIT; ++it) {
    const int kb = it * KB;
    if (it + 1 < NIT) stage((it + 1) & 1, kb + KB);
    if (kb + 2 * KB < NN) __builtin_prefetch(arow + kb + 2 * KB, 0, 1);

    const uint16_t* bbase = &Bs[it & 1][0];
#pragma unroll
    for (int ks = 0; ks < 2; ++ks) {
      const int k0 = kb + ks * 32;
      // 16 fp32 of this lane's A slice (two 32B chunks, per WMMA A layout)
      const float4 f0 = *(const float4*)(arow + k0 + g * 8);
      const float4 f1 = *(const float4*)(arow + k0 + g * 8 + 4);
      const float4 f2 = *(const float4*)(arow + k0 + 16 + g * 8);
      const float4 f3 = *(const float4*)(arow + k0 + 16 + g * 8 + 4);

      rowsum += f0.x + f0.y + f0.z + f0.w;
      rowsum += f1.x + f1.y + f1.z + f1.w;
      rowsum += f2.x + f2.y + f2.z + f2.w;
      rowsum += f3.x + f3.y + f3.z + f3.w;

      v16bf Av;
      Av[0]  = (__bf16)f0.x; Av[1]  = (__bf16)f0.y;
      Av[2]  = (__bf16)f0.z; Av[3]  = (__bf16)f0.w;
      Av[4]  = (__bf16)f1.x; Av[5]  = (__bf16)f1.y;
      Av[6]  = (__bf16)f1.z; Av[7]  = (__bf16)f1.w;
      Av[8]  = (__bf16)f2.x; Av[9]  = (__bf16)f2.y;
      Av[10] = (__bf16)f2.z; Av[11] = (__bf16)f2.w;
      Av[12] = (__bf16)f3.x; Av[13] = (__bf16)f3.y;
      Av[14] = (__bf16)f3.z; Av[15] = (__bf16)f3.w;

      // Preload all 8 B tiles, then burst the WMMAs (lets the scheduler use
      // decrementing s_wait_dscnt instead of wait-0 before every wmma).
      union BU { v16bf v; uint4 q[2]; };
      BU B[8];
#pragma unroll
      for (int dt = 0; dt < 8; ++dt) {
        const uint16_t* bp = bbase + (dt * 16 + ll) * BSTR + ks * 32 + g * 8;
        B[dt].q[0] = *(const uint4*)(bp);
        B[dt].q[1] = *(const uint4*)(bp + 16);
      }
#pragma unroll
      for (int dt = 0; dt < 8; ++dt)
        acc[dt] = __builtin_amdgcn_wmma_f32_16x16x32_bf16(
            false, Av, false, B[dt].v, (short)0, acc[dt], false, false);
    }
    __syncthreads();
  }

  // degree: lane L holds partial rowsum of row (L&15); combine halves
  rowsum += __shfl_xor(rowsum, 16, 32);
  const float deg = rowsum + 1.0f;
  float dinv[8];
#pragma unroll
  for (int i = 0; i < 8; ++i) {
    const float di = __shfl(deg, g * 8 + i, 32);
    dinv[i] = 1.0f / di;
  }

  // C layout: VGPR i -> row (i + 8*g), col = dt*16 + (lane&15)
#pragma unroll
  for (int dt = 0; dt < 8; ++dt)
#pragma unroll
    for (int i = 0; i < 8; ++i) {
      const int r = rowbase + g * 8 + i;
      aggbf[(size_t)r * DD + dt * 16 + ll] = (__bf16)(acc[dt][i] * dinv[i]);
    }
}

// ---------------------------------------------------------------------------
// Kernel 4: out = [features | agg] @ W^T  (K = 256, all bf16, fp32 accum).
// B(k=d, n=o) = W[o][d]; Wbf row-major [o][256] is already the B panel layout.
// 64 rows per WG (4 waves) -> 256 WGs.
// ---------------------------------------------------------------------------
__global__ __launch_bounds__(32 * NWAVE) void k_out_gemm(
    const __bf16* __restrict__ featbf, const __bf16* __restrict__ aggbf,
    const __bf16* __restrict__ wbf, float* __restrict__ out) {
  const int tid  = threadIdx.x;
  const int wave = tid >> 5;
  const int lane = tid & 31;
  const int ll   = lane & 15;
  const int g    = lane >> 4;
  const int rowbase = blockIdx.x * (16 * NWAVE) + wave * 16;
  const int row     = rowbase + ll;

  v8f acc[8];
#pragma unroll
  for (int i = 0; i < 8; ++i)
#pragma unroll
    for (int j = 0; j < 8; ++j) acc[i][j] = 0.0f;

#pragma unroll
  for (int kt = 0; kt < 8; ++kt) {
    const int kb = kt * 32;
    const uint16_t* abase = (kt < 4)
        ? ((const uint16_t*)featbf + (size_t)row * DD + kb)
        : ((const uint16_t*)aggbf + (size_t)row * DD + (kb - 128));
    union { v16bf v; uint4 q[2]; } A;
    A.q[0] = *(const uint4*)(abase + g * 8);
    A.q[1] = *(const uint4*)(abase + 16 + g * 8);

#pragma unroll
    for (int ot = 0; ot < 8; ++ot) {
      const uint16_t* bp =
          (const uint16_t*)wbf + (size_t)(ot * 16 + ll) * 256 + kb + g * 8;
      union { v16bf v; uint4 q[2]; } B;
      B.q[0] = *(const uint4*)(bp);
      B.q[1] = *(const uint4*)(bp + 16);
      acc[ot] = __builtin_amdgcn_wmma_f32_16x16x32_bf16(
          false, A.v, false, B.v, (short)0, acc[ot], false, false);
    }
  }

#pragma unroll
  for (int ot = 0; ot < 8; ++ot)
#pragma unroll
    for (int i = 0; i < 8; ++i)
      out[(size_t)(rowbase + g * 8 + i) * DD + ot * 16 + ll] = acc[ot][i];
}

// ---------------------------------------------------------------------------
extern "C" void kernel_launch(void* const* d_in, const int* in_sizes, int n_in,
                              void* d_out, int out_size, void* d_ws, size_t ws_size,
                              hipStream_t stream) {
  const float* adj   = (const float*)d_in[0];  // [N, N]
  const float* feat  = (const float*)d_in[1];  // [N, 128]
  const float* neigh = (const float*)d_in[2];  // [N, 128]
  const float* W     = (const float*)d_in[3];  // [128, 256]
  float* out = (float*)d_out;

  char* ws = (char*)d_ws;
  const size_t SZ = (size_t)NN * DD * sizeof(uint16_t);  // 4 MiB each
  __bf16* neighT = (__bf16*)(ws);
  __bf16* featbf = (__bf16*)(ws + SZ);
  __bf16* aggbf  = (__bf16*)(ws + 2 * SZ);
  __bf16* wbf    = (__bf16*)(ws + 3 * SZ);  // 64 KiB

  k_transpose_bf16<<<dim3(NN / 32, DD / 32), dim3(32, 8), 0, stream>>>(neigh, neighT);

  const int total = NN * DD + DD * 2 * DD;
  k_cvt_bf16<<<(total + 255) / 256, 256, 0, stream>>>(feat, W, featbf, wbf);

  k_agg_gemm<<<NN / (16 * NWAVE), 32 * NWAVE, 0, stream>>>(adj, neighT, aggbf);

  k_out_gemm<<<NN / (16 * NWAVE), 32 * NWAVE, 0, stream>>>(featbf, aggbf, wbf, out);
}
